// CMCloss_55473797595622
// MI455X (gfx1250) — matrix operand order repaired and checked by reading, not compiled, not run on hardware
//
#include <hip/hip_runtime.h>
#include <hip/hip_bf16.h>
#include <math.h>

#define B_N 2048
#define D_N 8
#define F_N 256
#define INV_T 2.0f     // 1 / temperature (0.5)
#define TILE 128
#define KC   32
#define LDP  264       // padded LDS row stride in ushorts: 512B data + 16B pad = 528B (16B-aligned, bank-friendly)

typedef __attribute__((ext_vector_type(16))) __bf16 v16bf;
typedef __attribute__((ext_vector_type(8)))  float  v8f;
typedef unsigned int u32x4 __attribute__((ext_vector_type(4)));
typedef int          i32x4 __attribute__((ext_vector_type(4)));
typedef int          i32x8 __attribute__((ext_vector_type(8)));

union FragU {
    v16bf v;
    uint4 q[2];
};

// ---------------- 1) normalize rows, emit bf16 copy in [d][b][f] layout ----------------
__global__ void nrm_k(const float* __restrict__ z, unsigned short* __restrict__ znb,
                      float* __restrict__ scales) {
    __shared__ float red[256];
    __shared__ float sc;
    int row = blockIdx.x;            // row = b*8 + d
    int t   = threadIdx.x;           // 256 threads, one per feature
    float x = z[row * F_N + t];
    red[t] = x * x;
    __syncthreads();
    for (int s = 128; s > 0; s >>= 1) {
        if (t < s) red[t] += red[t + s];
        __syncthreads();
    }
    if (t == 0) {
        float norm = sqrtf(red[0]);
        float sv = 1.0f / fmaxf(norm, 1e-12f);
        sc = sv;
        scales[row] = sv;
    }
    __syncthreads();
    float v = x * sc;
    unsigned u = __float_as_uint(v);                       // RNE f32 -> bf16
    unsigned r = (u + 0x7FFFu + ((u >> 16) & 1u)) >> 16;
    int b = row >> 3, d = row & 7;
    znb[((size_t)d * B_N + b) * F_N + t] = (unsigned short)r;
}

// ---------------- 2) zero the negative-term accumulator ----------------
__global__ void zero_k(float* __restrict__ p, int n) {
    int i = blockIdx.x * blockDim.x + threadIdx.x;
    if (i < n) p[i] = 0.0f;
}

// ---------------- 3) positive term: per-sample 8x8 gram (tiny) ----------------
__global__ void pos_k(const float* __restrict__ z, const float* __restrict__ scales,
                      float* __restrict__ pos) {
    __shared__ float zs[D_N * F_N];
    __shared__ float sv[64];
    int b = blockIdx.x;
    int t = threadIdx.x;             // 64 threads
    for (int i = 0; i < 32; ++i) {
        int idx = t + 64 * i;
        int d = idx >> 8;
        zs[idx] = z[(size_t)b * (D_N * F_N) + idx] * scales[b * D_N + d];
    }
    __syncthreads();
    int d = t >> 3, e = t & 7;
    float dot = 0.0f;
    #pragma unroll 8
    for (int f = 0; f < F_N; ++f) dot += zs[d * F_N + f] * zs[e * F_N + f];
    sv[t] = (d == e) ? 0.0f : __expf(dot * INV_T);
    __syncthreads();
    if (t == 0) {
        float s = 0.0f;
        for (int i = 0; i < 64; ++i) s += sv[i];
        pos[b] = s;
    }
}

// -------- TDM helper: DMA a 16-row x 256-col bf16 slice into LDS with HW row padding --------
#if __has_builtin(__builtin_amdgcn_tensor_load_to_lds)
#define HAVE_TDM 1
__device__ __forceinline__ void tdm_load_slice(unsigned lds_addr, const unsigned short* gsrc) {
    unsigned long long ga = (unsigned long long)(size_t)gsrc;
    u32x4 g0;
    g0[0] = 1u;                                        // count=1 (valid user descriptor)
    g0[1] = lds_addr;                                  // LDS byte address
    g0[2] = (unsigned)(ga & 0xFFFFFFFFu);              // global_addr[31:0]
    g0[3] = (unsigned)(ga >> 32) | (2u << 30);         // global_addr[56:32] | type=2 ("image")
    i32x8 g1;
    g1[0] = (int)((1u << 16)      // data_size = 2 bytes
                | (1u << 20)      // pad_enable
                | (6u << 22)      // pad_interval: 128 DWORDs (= one 512B row)
                | (3u << 25));    // pad_amount: 4 DWORDs (16B) -> 528B LDS row stride
    g1[1] = (int)(256u << 16);    // tensor_dim0[15:0] = 256
    g1[2] = (int)(2048u << 16);   // tensor_dim0 hi = 0 | tensor_dim1[15:0] = 2048
    g1[3] = (int)(256u << 16);    // tensor_dim1 hi = 0 | tile_dim0 = 256
    g1[4] = 16;                   // tile_dim1 = 16, tile_dim2 = 0 (2-D tile)
    g1[5] = 256;                  // tensor_dim0_stride = 256 elements
    g1[6] = 0;
    g1[7] = 0;
    i32x4 zz  = {0, 0, 0, 0};                // groups 2/3 unused for 2-D tiles
    i32x8 zz8 = {0, 0, 0, 0, 0, 0, 0, 0};    // extra operand in 6-arg form: NULL descriptor
    __builtin_amdgcn_tensor_load_to_lds(g0, g1, zz, zz, zz8, 0);
}
#else
#define HAVE_TDM 0
#endif

// ---------------- 4) negative term: per-view 2048x2048 gram via WMMA bf16 ----------------
// grid (16,16,8), block 256 (= 8 wave32). Each block: 128x128 tile of view d.
// Whole 128x256 A and B panels staged in LDS (TDM, one DMA per wave per panel slice).
__global__ void neg_k(const unsigned short* __restrict__ znb, float* __restrict__ negsum) {
    extern __shared__ unsigned short smem[];           // 2 * 128 * LDP ushorts
    unsigned short* At = smem;
    unsigned short* Bt = smem + TILE * LDP;

    int d       = blockIdx.z;
    int tileRow = blockIdx.y * TILE;
    int tileCol = blockIdx.x * TILE;
    const unsigned short* Zd = znb + (size_t)d * B_N * F_N;

    int t    = threadIdx.x;
    int lane = t & 31;
    int w    = t >> 5;               // wave id 0..7 -> A rows [tileRow + 16w, +16)

#if HAVE_TDM
    {
        unsigned wu = __builtin_amdgcn_readfirstlane((unsigned)threadIdx.x) >> 5;  // SGPR wave id
        unsigned ldsA = (unsigned)(size_t)At + wu * 16u * (LDP * 2u);
        unsigned ldsB = (unsigned)(size_t)Bt + wu * 16u * (LDP * 2u);
        tdm_load_slice(ldsA, Zd + (size_t)(tileRow + (int)wu * 16) * F_N);
        tdm_load_slice(ldsB, Zd + (size_t)(tileCol + (int)wu * 16) * F_N);
        __builtin_amdgcn_s_wait_tensorcnt(0);
    }
#else
    for (int p = 0; p < 8; ++p) {                      // cooperative fallback: 2048 32B chunks/panel
        int c   = t + 256 * p;
        int r   = c >> 4;
        int col = (c & 15) << 4;
        const uint4* ga = (const uint4*)(Zd + (size_t)(tileRow + r) * F_N + col);
        uint4* la = (uint4*)(At + r * LDP + col);
        la[0] = ga[0]; la[1] = ga[1];
        const uint4* gb = (const uint4*)(Zd + (size_t)(tileCol + r) * F_N + col);
        uint4* lb = (uint4*)(Bt + r * LDP + col);
        lb[0] = gb[0]; lb[1] = gb[1];
    }
#endif
    __syncthreads();

    v8f acc[8] = {};                 // 8 col-tiles of 16x16 f32

    // A-fragment: lane -> row (lane&15), K-chunks at {kh, kh+16}, kh = 0|8
    int fr = lane & 15;
    int kh = (lane < 16) ? 0 : 8;
    // B-fragment: lane -> col (lane&15), K-chunks at {kb, kb+16}, kb = 0|8
    int kb = (lane < 16) ? 0 : 8;

    const unsigned short* Abase = At + (w * 16 + fr) * LDP;
    #pragma unroll
    for (int kc = 0; kc < F_N; kc += KC) {
        FragU af;
        af.q[0] = *(const uint4*)(Abase + kc + kh);
        af.q[1] = *(const uint4*)(Abase + kc + kh + 16);
        #pragma unroll
        for (int ct = 0; ct < 8; ++ct) {
            const unsigned short* Bbase = Bt + (ct * 16 + fr) * LDP + kc + kb;
            FragU bfr;
            bfr.q[0] = *(const uint4*)(Bbase);
            bfr.q[1] = *(const uint4*)(Bbase + 16);
            acc[ct] = __builtin_amdgcn_wmma_f32_16x16x32_bf16(
                false, af.v, false, bfr.v, (short)0, acc[ct], false, false);
        }
    }

    // C layout: VGPR v, lane L -> (M = v + (L>=16?8:0), N = L&15)
    int mo = (lane < 16) ? 0 : 8;
    int nn = lane & 15;
    float rowsum[8];
    #pragma unroll
    for (int v = 0; v < 8; ++v) rowsum[v] = 0.0f;
    #pragma unroll
    for (int ct = 0; ct < 8; ++ct) {
        int gcol = tileCol + ct * 16 + nn;
        #pragma unroll
        for (int v = 0; v < 8; ++v) {
            int grow = tileRow + w * 16 + v + mo;
            float e = __expf(acc[ct][v] * INV_T);
            rowsum[v] += (grow == gcol) ? 0.0f : e;
        }
    }
    #pragma unroll
    for (int v = 0; v < 8; ++v) {
        float s = rowsum[v];
        s += __shfl_xor(s, 1, 32);
        s += __shfl_xor(s, 2, 32);
        s += __shfl_xor(s, 4, 32);
        s += __shfl_xor(s, 8, 32);
        rowsum[v] = s;
    }
    if (nn == 0) {
        #pragma unroll
        for (int v = 0; v < 8; ++v)
            atomicAdd(&negsum[tileRow + w * 16 + mo + v], rowsum[v]);
    }
}

// ---------------- 5) logits + (-sum log_softmax)/B = LSE - mean ----------------
__global__ void fin_k(const float* __restrict__ pos, const float* __restrict__ neg,
                      float* __restrict__ out) {
    __shared__ float red[256];
    int t = threadIdx.x;
    float l[8];
    float mx = -1e30f;
    #pragma unroll
    for (int i = 0; i < 8; ++i) {
        int b = t * 8 + i;
        float p = pos[b];
        float n = neg[b] * (1.0f / (float)(B_N - 1));
        l[i] = p / (p + n);
        mx = fmaxf(mx, l[i]);
    }
    red[t] = mx; __syncthreads();
    for (int s = 128; s > 0; s >>= 1) { if (t < s) red[t] = fmaxf(red[t], red[t + s]); __syncthreads(); }
    mx = red[0]; __syncthreads();
    float se = 0.0f, su = 0.0f;
    #pragma unroll
    for (int i = 0; i < 8; ++i) { se += __expf(l[i] - mx); su += l[i]; }
    red[t] = se; __syncthreads();
    for (int s = 128; s > 0; s >>= 1) { if (t < s) red[t] += red[t + s]; __syncthreads(); }
    se = red[0]; __syncthreads();
    red[t] = su; __syncthreads();
    for (int s = 128; s > 0; s >>= 1) { if (t < s) red[t] += red[t + s]; __syncthreads(); }
    if (t == 0) {
        su = red[0];
        out[0] = mx + logf(se) - su / (float)B_N;
    }
}

extern "C" void kernel_launch(void* const* d_in, const int* in_sizes, int n_in,
                              void* d_out, int out_size, void* d_ws, size_t ws_size,
                              hipStream_t stream) {
    (void)in_sizes; (void)n_in; (void)out_size; (void)ws_size;
    const float* z = (const float*)d_in[0];
    float* out = (float*)d_out;

    char* ws = (char*)d_ws;
    unsigned short* znb = (unsigned short*)ws;                      // 8 MiB bf16 zn, [d][b][f]
    float* scales = (float*)(ws + (size_t)B_N * D_N * F_N * 2);     // 64 KiB
    float* pos    = scales + B_N * D_N;                             // 8 KiB
    float* neg    = pos + B_N;                                      // 8 KiB

    nrm_k <<<B_N * D_N, 256, 0, stream>>>(z, znb, scales);
    zero_k<<<(B_N + 255) / 256, 256, 0, stream>>>(neg, B_N);
    pos_k <<<B_N, 64, 0, stream>>>(z, scales, pos);
    dim3 grid(B_N / TILE, B_N / TILE, D_N);
    size_t lds_bytes = (size_t)2 * TILE * LDP * sizeof(unsigned short);  // ~135 KB of the 320 KB WGP LDS
    neg_k <<<grid, 256, lds_bytes, stream>>>(znb, neg);
    fin_k <<<1, 256, 0, stream>>>(pos, neg, out);
}
